// SelfAttention_19378892439617
// MI455X (gfx1250) — matrix operand order repaired and checked by reading, not compiled, not run on hardware
//
#include <hip/hip_runtime.h>

typedef _Float16 v16h __attribute__((ext_vector_type(16)));
typedef _Float16 v8h  __attribute__((ext_vector_type(8)));
typedef _Float16 v4h  __attribute__((ext_vector_type(4)));
typedef float    v8f  __attribute__((ext_vector_type(8)));
typedef unsigned int u32x4 __attribute__((ext_vector_type(4)));
typedef int          i32x8 __attribute__((ext_vector_type(8)));
typedef int          i32x4 __attribute__((ext_vector_type(4)));

static __device__ __forceinline__ v16h load_frag(const _Float16* p0, const _Float16* p1) {
    v16h r;
    *reinterpret_cast<v8h*>(&r)       = *reinterpret_cast<const v8h*>(p0);
    *(reinterpret_cast<v8h*>(&r) + 1) = *reinterpret_cast<const v8h*>(p1);
    return r;
}

static __device__ __forceinline__ v8f wmma_f16(v16h a, v16h b, v8f c) {
    // D = A(16x32 f16) * B(32x16 f16) + C(16x16 f32)
    return __builtin_amdgcn_wmma_f32_16x16x32_f16(false, a, false, b, (short)0, c, false, false);
}

// ---------------------------------------------------------------------------
// TDM: async 2D tile load (global -> LDS), 16-bit elements.
// Descriptor packing per CDNA5 ISA.
//   pad codes: interval = 2^(code+1) DWORDs, amount = (code+1) DWORDs.
// ---------------------------------------------------------------------------
static __device__ __forceinline__ void tdm_load_2d_f16(
    const void* gptr, unsigned lds_byte_off,
    unsigned tile_d0, unsigned tile_d1,
    unsigned tensor_d0, unsigned tensor_d1,
    unsigned long long stride0_elems,
    unsigned pad_interval_code, unsigned pad_amount_code, unsigned pad_en)
{
    unsigned long long ga = (unsigned long long)(size_t)gptr;
    u32x4 g0;
    g0[0] = 1u;                                            // count=1, user flags 0
    g0[1] = lds_byte_off;                                  // lds_addr [63:32]
    g0[2] = (unsigned)(ga & 0xFFFFFFFFull);                // global_addr lo
    g0[3] = (unsigned)((ga >> 32) & 0x01FFFFFFull) | (2u << 30);  // addr hi | type=2
    unsigned w0 = (1u << 16)                               // data_size=1 (2 bytes)
                | (pad_en << 20)
                | (pad_interval_code << 22)
                | (pad_amount_code << 25);
    unsigned w1 = (tensor_d0 & 0xFFFFu) << 16;             // tensor_dim0[15:0] @ bits 63:48
    unsigned w2 = ((tensor_d0 >> 16) & 0xFFFFu) | ((tensor_d1 & 0xFFFFu) << 16);
    unsigned w3 = ((tensor_d1 >> 16) & 0xFFFFu) | ((tile_d0 & 0xFFFFu) << 16);
    unsigned w4 = (tile_d1 & 0xFFFFu);                     // tile_dim2 = 0 (2D)
    unsigned w5 = (unsigned)(stride0_elems & 0xFFFFFFFFull);
    unsigned w6 = (unsigned)((stride0_elems >> 32) & 0xFFFFull);  // dim1_stride unused
    i32x8 g1;
    g1[0] = (int)w0; g1[1] = (int)w1; g1[2] = (int)w2; g1[3] = (int)w3;
    g1[4] = (int)w4; g1[5] = (int)w5; g1[6] = (int)w6; g1[7] = 0;
    i32x4 z4; z4[0] = 0; z4[1] = 0; z4[2] = 0; z4[3] = 0;
#if defined(__clang_major__) && (__clang_major__ >= 23)
    i32x8 z8; z8[0]=0; z8[1]=0; z8[2]=0; z8[3]=0; z8[4]=0; z8[5]=0; z8[6]=0; z8[7]=0;
    __builtin_amdgcn_tensor_load_to_lds(g0, g1, z4, z4, z8, 0);
#else
    __builtin_amdgcn_tensor_load_to_lds(g0, g1, z4, z4, 0);
#endif
}

// ---------------------------------------------------------------------------
// Kernel 1: fp32 -> f16 cast (weights), 4 elems/thread, sizes exact.
// ---------------------------------------------------------------------------
__global__ __launch_bounds__(256) void cast_f32_to_f16_x4(const float* __restrict__ in,
                                                          _Float16* __restrict__ out) {
    int i = (blockIdx.x * 256 + threadIdx.x) * 4;
    float4 v = *reinterpret_cast<const float4*>(in + i);
    v4h h;
    h[0] = (_Float16)v.x; h[1] = (_Float16)v.y; h[2] = (_Float16)v.z; h[3] = (_Float16)v.w;
    *reinterpret_cast<v4h*>(out + i) = h;
}

// ---------------------------------------------------------------------------
// Kernel 2/4: Y[m][n] = sum_k A[m][k] * W[n][k]   (A: 8192x1024, W: 1024x1024)
// 256 threads = 8 waves; block tile 256(M) x 64(N); each wave 32x64; K-chunks 32.
// Double-buffered: W-tile via TDM (pad -> LDS stride 40 halves), A-tile manual;
// next chunk's loads are issued before computing the current chunk. Loop is
// unrolled x2 so the buffer index is a compile-time constant.
// A_F32: A fp32 (cast into LDS) else f16. OUT_HEAD: f16 [b][h][s][64] else fp32.
// ---------------------------------------------------------------------------
template <int A_F32, int OUT_HEAD>
__global__ __launch_bounds__(256) void gemm256(const void* __restrict__ Ap,
                                               const _Float16* __restrict__ Wh,
                                               void* __restrict__ Outp) {
    __shared__ __align__(16) _Float16 As[2][256 * 40];  // [row][k], stride 40 halves
    __shared__ __align__(16) _Float16 Bs[2][64 * 40];   // [n][k], stride 40 halves (TDM pad)

    const int tid  = threadIdx.x;
    const int lane = tid & 31;
    const int w    = tid >> 5;
    const int lm   = lane & 15;
    const int hi   = (lane >> 4) & 1;
    const int m0   = blockIdx.y * 256;
    const int n0   = blockIdx.x * 64;
    const int abase = hi ? 8 : 0;
    const int kofs  = hi ? 16 : 0;

    // Issue all loads (TDM W-tile + manual A-tile) for K-chunk `kc` into buffer `buf`.
    auto issue_tile = [&](int kc, int buf) {
        if (w == 0) {
            tdm_load_2d_f16(Wh + (size_t)n0 * 1024 + kc, (unsigned)(size_t)Bs[buf],
                            /*tile*/ 32, 64, /*tensor*/ 32, 64,
                            /*stride0*/ 1024ull,
                            /*interval 2^(3+1)=16DW*/ 3, /*amount 4DW*/ 3, 1);
        }
        if (A_F32) {
            const float* A = (const float*)Ap;
#pragma unroll
            for (int j = 0; j < 8; j++) {
                int idx = tid + j * 256;               // 2048 float4's
                int fb = idx * 4;
                int row = fb >> 5, col = fb & 31;
                float4 v = *reinterpret_cast<const float4*>(A + (size_t)(m0 + row) * 1024 + kc + col);
                v4h hv;
                hv[0] = (_Float16)v.x; hv[1] = (_Float16)v.y;
                hv[2] = (_Float16)v.z; hv[3] = (_Float16)v.w;
                *reinterpret_cast<v4h*>(&As[buf][row * 40 + col]) = hv;
            }
        } else {
            const _Float16* A = (const _Float16*)Ap;
#pragma unroll
            for (int j = 0; j < 4; j++) {
                int idx = tid + j * 256;               // 1024 v8h's
                int hb = idx * 8;
                int row = hb >> 5, col = hb & 31;
                *reinterpret_cast<v8h*>(&As[buf][row * 40 + col]) =
                    *reinterpret_cast<const v8h*>(A + (size_t)(m0 + row) * 1024 + kc + col);
            }
        }
    };

    v8f acc[2][4];
#pragma unroll
    for (int s = 0; s < 2; s++)
#pragma unroll
        for (int t = 0; t < 4; t++)
#pragma unroll
            for (int e = 0; e < 8; e++) acc[s][t][e] = 0.0f;

    // Prologue: fill buffer 0.
    issue_tile(0, 0);
    __builtin_amdgcn_s_wait_tensorcnt(0);
    __syncthreads();

#pragma unroll 2
    for (int it = 0; it < 32; it++) {
        const int cur = it & 1;
        if (it + 1 < 32) issue_tile((it + 1) * 32, cur ^ 1);  // overlap with compute

        const _Float16* ap0 = &As[cur][(32 * w + lm) * 40];
        const _Float16* ap1 = &As[cur][(32 * w + 16 + lm) * 40];
        v16h a0 = load_frag(ap0 + abase, ap0 + 16 + abase);
        v16h a1 = load_frag(ap1 + abase, ap1 + 16 + abase);
#pragma unroll
        for (int t = 0; t < 4; t++) {
            const _Float16* bp = &Bs[cur][(16 * t + lm) * 40 + abase];
            v16h b = load_frag(bp, bp + 16);
            acc[0][t] = wmma_f16(a0, b, acc[0][t]);
            acc[1][t] = wmma_f16(a1, b, acc[1][t]);
        }

        __builtin_amdgcn_s_wait_tensorcnt(0);   // next buffer's DMA complete
        __syncthreads();                        // next buffer's stores visible
    }

    if (OUT_HEAD) {
        _Float16* Out = (_Float16*)Outp;
#pragma unroll
        for (int s2 = 0; s2 < 2; s2++)
#pragma unroll
            for (int t = 0; t < 4; t++) {
                int n = n0 + 16 * t + lm;
                int hh = n >> 6, d = n & 63;
#pragma unroll
                for (int i = 0; i < 8; i++) {
                    int m = m0 + 32 * w + 16 * s2 + i + 8 * hi;
                    int bb = m >> 11, s = m & 2047;
                    Out[(((size_t)bb * 16 + hh) * 2048 + s) * 64 + d] = (_Float16)acc[s2][t][i];
                }
            }
    } else {
        float* Out = (float*)Outp;
#pragma unroll
        for (int s2 = 0; s2 < 2; s2++)
#pragma unroll
            for (int t = 0; t < 4; t++) {
                int n = n0 + 16 * t + lm;
#pragma unroll
                for (int i = 0; i < 8; i++) {
                    int m = m0 + 32 * w + 16 * s2 + i + 8 * hi;
                    Out[(size_t)m * 1024 + n] = acc[s2][t][i];
                }
            }
    }
}

// ---------------------------------------------------------------------------
// Kernel 3: flash attention. Block = (b,h, 64 q-rows); 4 waves x 16 q-rows.
// Double-buffered: K tile via TDM (pad -> stride 72 halves), V transposed
// manually, mask staged through LDS; next block's loads are issued before
// computing the current one. Score WMMAs are batched (4 independent
// accumulators) before any VALU consumption to fill WMMA->VALU hazard slots.
// ---------------------------------------------------------------------------
__global__ __launch_bounds__(128) void flash64(const _Float16* __restrict__ Qh,
                                               const _Float16* __restrict__ Kh,
                                               const _Float16* __restrict__ Vh,
                                               const unsigned char* __restrict__ maskp,
                                               _Float16* __restrict__ Oh) {
    __shared__ __align__(16) _Float16 Ks[2][64 * 72];      // [kk][d], stride 72 (TDM pad)
    __shared__ __align__(16) _Float16 Vt[2][64 * 72];      // [d][kk], stride 72 (transposed)
    __shared__ __align__(16) unsigned char Ms[2][64 * 64]; // mask tile [q][kk]
    __shared__ __align__(16) _Float16 Ps[4][16 * 72];      // per-wave P tile [q][kk]

    const int tid  = threadIdx.x;
    const int lane = tid & 31;
    const int w    = tid >> 5;
    const int lm   = lane & 15;
    const int hi   = (lane >> 4) & 1;
    const int qb   = blockIdx.x;          // 0..31
    const int bh   = blockIdx.y;          // 0..63
    const int b    = bh >> 4;
    const int h    = bh & 15;
    const int q0   = qb * 64;
    const int abase = hi ? 8 : 0;
    const int kofs  = hi ? 16 : 0;
    const float scale = 0.125f;           // 1/sqrt(64)

    const unsigned char* mbase = maskp + ((size_t)b * 2048 + q0) * 2048;

    // Issue all loads (TDM K, manual V-transpose, mask stage) for block `kbb`.
    auto issue_tile = [&](int kbb, int buf) {
        const int kk0 = kbb * 64;
        const _Float16* kg = Kh + ((size_t)bh * 2048 + kk0) * 64;  // 4096 contiguous halves
        const _Float16* vg = Vh + ((size_t)bh * 2048 + kk0) * 64;
        if (w == 0) {
            tdm_load_2d_f16(kg, (unsigned)(size_t)Ks[buf],
                            /*tile*/ 64, 64, /*tensor*/ 64, 64,
                            /*stride0*/ 64ull,
                            /*interval 2^(4+1)=32DW*/ 4, /*amount 4DW*/ 3, 1);
        }
#pragma unroll
        for (int j = 0; j < 4; j++) {
            int idx = tid + j * 128;       // 512 v8h's
            int hb = idx * 8;
            int row = hb >> 6, col = hb & 63;
            v8h vv = *reinterpret_cast<const v8h*>(vg + hb);
#pragma unroll
            for (int e = 0; e < 8; e++) Vt[buf][(col + e) * 72 + row] = vv[e];
        }
#pragma unroll
        for (int j = 0; j < 2; j++) {
            int idx = tid + j * 128;       // 256 uint4's
            int byi = idx * 16;
            int row = byi >> 6, col = byi & 63;
            *reinterpret_cast<uint4*>(&Ms[buf][row * 64 + col]) =
                *reinterpret_cast<const uint4*>(mbase + (size_t)row * 2048 + kk0 + col);
        }
    };

    // Q fragments live in registers for the whole kernel
    const int qrow = q0 + 16 * w + lm;
    const _Float16* qp = Qh + ((size_t)bh * 2048 + qrow) * 64;
    v16h qa0 = load_frag(qp + abase,      qp + 16 + abase);
    v16h qa1 = load_frag(qp + 32 + abase, qp + 48 + abase);

    v8f acc[4];
#pragma unroll
    for (int t = 0; t < 4; t++)
#pragma unroll
        for (int e = 0; e < 8; e++) acc[t][e] = 0.0f;
    float mi[8], li[8];
#pragma unroll
    for (int i = 0; i < 8; i++) { mi[i] = -1e30f; li[i] = 0.0f; }

    // Prologue: fill buffer 0.
    issue_tile(0, 0);
    __builtin_amdgcn_s_wait_tensorcnt(0);
    __syncthreads();

#pragma unroll 2
    for (int kb = 0; kb < 32; kb++) {
        const int cur = kb & 1;
        if (kb + 1 < 32) issue_tile(kb + 1, cur ^ 1);   // overlap with compute

        // ---- score WMMAs: 8 matrix ops into 4 independent accumulators ----
        v8f sv[4];
#pragma unroll
        for (int t = 0; t < 4; t++) {
            const _Float16* bp = &Ks[cur][(16 * t + lm) * 72];
            v16h b0 = load_frag(bp + abase,      bp + abase + 16);
            v16h b1 = load_frag(bp + 32 + abase, bp + 32 + abase + 16);
            v8f z;
#pragma unroll
            for (int e = 0; e < 8; e++) z[e] = 0.0f;
            z = wmma_f16(qa0, b0, z);
            sv[t] = wmma_f16(qa1, b1, z);
        }

        // ---- scale + mask (separate pass -> hazard slots filled by WMMAs) ----
        float sc[4][8];
#pragma unroll
        for (int t = 0; t < 4; t++)
#pragma unroll
            for (int i = 0; i < 8; i++) {
                float v = sv[t][i] * scale;
                if (Ms[cur][(16 * w + i + 8 * hi) * 64 + 16 * t + lm]) v = -1e7f;
                sc[t][i] = v;
            }

        // ---- online softmax update (row stats across 16-lane halves) ----
#pragma unroll
        for (int i = 0; i < 8; i++) {
            float rmax = fmaxf(fmaxf(sc[0][i], sc[1][i]), fmaxf(sc[2][i], sc[3][i]));
#pragma unroll
            for (int o = 1; o < 16; o <<= 1) rmax = fmaxf(rmax, __shfl_xor(rmax, o, 32));
            float mnew  = fmaxf(mi[i], rmax);
            float alpha = __expf(mi[i] - mnew);
            float rsum = 0.0f;
#pragma unroll
            for (int t = 0; t < 4; t++) {
                float p = __expf(sc[t][i] - mnew);
                Ps[w][(i + 8 * hi) * 72 + 16 * t + lm] = (_Float16)p;
                rsum += p;
            }
#pragma unroll
            for (int o = 1; o < 16; o <<= 1) rsum += __shfl_xor(rsum, o, 32);
            li[i] = li[i] * alpha + rsum;
            mi[i] = mnew;
#pragma unroll
            for (int t = 0; t < 4; t++) acc[t][i] *= alpha;
        }

        // ---- acc += P * V ----
#pragma unroll
        for (int c2 = 0; c2 < 2; c2++) {
            const _Float16* pp = &Ps[w][lm * 72 + 32 * c2 + abase];
            v16h pa = load_frag(pp, pp + 16);
#pragma unroll
            for (int td = 0; td < 4; td++) {
                const _Float16* vp = &Vt[cur][(16 * td + lm) * 72 + 32 * c2 + abase];
                v16h vb = load_frag(vp, vp + 16);
                acc[td] = wmma_f16(pa, vb, acc[td]);
            }
        }

        __builtin_amdgcn_s_wait_tensorcnt(0);   // next buffer's DMA complete
        __syncthreads();                        // next buffer's stores visible
    }

    // ---- epilogue: O[b][s][h*64+d] = acc / l ----
#pragma unroll
    for (int td = 0; td < 4; td++) {
        int d = 16 * td + lm;
#pragma unroll
        for (int i = 0; i < 8; i++) {
            int srow = q0 + 16 * w + i + 8 * hi;
            Oh[((size_t)b * 2048 + srow) * 1024 + h * 64 + d] = (_Float16)(acc[td][i] / li[i]);
        }
    }
}

// ---------------------------------------------------------------------------
extern "C" void kernel_launch(void* const* d_in, const int* in_sizes, int n_in,
                              void* d_out, int out_size, void* d_ws, size_t ws_size,
                              hipStream_t stream) {
    (void)in_sizes; (void)n_in; (void)out_size; (void)ws_size;
    const float* x  = (const float*)d_in[0];
    const float* Wq = (const float*)d_in[1];
    const float* Wk = (const float*)d_in[2];
    const float* Wv = (const float*)d_in[3];
    const float* Wo = (const float*)d_in[4];
    const unsigned char* mask = (const unsigned char*)d_in[5];

    char* ws = (char*)d_ws;
    const size_t WB = (size_t)1024 * 1024 * 2;   // 2 MB per f16 weight
    const size_t QB = (size_t)8192 * 1024 * 2;   // 16 MB per f16 activation
    _Float16* WqH = (_Float16*)(ws + 0 * WB);
    _Float16* WkH = (_Float16*)(ws + 1 * WB);
    _Float16* WvH = (_Float16*)(ws + 2 * WB);
    _Float16* WoH = (_Float16*)(ws + 3 * WB);
    _Float16* Qh  = (_Float16*)(ws + 4 * WB + 0 * QB);
    _Float16* Kh  = (_Float16*)(ws + 4 * WB + 1 * QB);
    _Float16* Vh  = (_Float16*)(ws + 4 * WB + 2 * QB);
    _Float16* Oh  = (_Float16*)(ws + 4 * WB + 3 * QB);

    // 1) cast weights
    cast_f32_to_f16_x4<<<1024, 256, 0, stream>>>(Wq, WqH);
    cast_f32_to_f16_x4<<<1024, 256, 0, stream>>>(Wk, WkH);
    cast_f32_to_f16_x4<<<1024, 256, 0, stream>>>(Wv, WvH);
    cast_f32_to_f16_x4<<<1024, 256, 0, stream>>>(Wo, WoH);

    // 2) Q/K/V projections (head-split f16 output)
    dim3 gg(16, 32), gb(256);
    gemm256<1, 1><<<gg, gb, 0, stream>>>(x, WqH, Qh);
    gemm256<1, 1><<<gg, gb, 0, stream>>>(x, WkH, Kh);
    gemm256<1, 1><<<gg, gb, 0, stream>>>(x, WvH, Vh);

    // 3) flash attention
    flash64<<<dim3(32, 64), 128, 0, stream>>>(Qh, Kh, Vh, mask, Oh);

    // 4) output projection (fp32 output)
    gemm256<0, 0><<<gg, gb, 0, stream>>>(Oh, WoH, (void*)d_out);
}
